// DecoderModel_41592463294953
// MI455X (gfx1250) — matrix operand order repaired and checked by reading, not compiled, not run on hardware
//
#include <hip/hip_runtime.h>

#define HIDDEN   256
#define SEQLEN   256
#define OUT_N    256
#define T_ENC    64
#define M_TILE   16      // batch rows per workgroup
#define NBLOCKS  32      // 512 / 16
#define NTHREADS 1024    // 32 waves

typedef __attribute__((ext_vector_type(16))) __bf16   v16bf;
typedef __attribute__((ext_vector_type(8)))  float    v8f;
typedef __attribute__((ext_vector_type(4)))  unsigned u4;
typedef __attribute__((ext_vector_type(2)))  unsigned u2;
typedef __attribute__((ext_vector_type(4)))  float    f4;
typedef unsigned short ushort_t;

union BF16Tile { u4 q[2]; v16bf v; };

__device__ __forceinline__ ushort_t f2bf(float f) {
    unsigned u = __float_as_uint(f);
    unsigned r = (u + 0x7FFFu + ((u >> 16) & 1u)) >> 16;   // RNE
    return (ushort_t)r;
}

__device__ __forceinline__ float fast_sigmoid(float x) {
    return 1.f / (1.f + __expf(-x));
}
__device__ __forceinline__ float fast_tanh(float x) {
    // tanh(x) = 1 - 2/(1 + exp(2x)) : straight-line v_exp/v_rcp path
    return 1.f - 2.f / (1.f + __expf(2.f * x));
}

__device__ __forceinline__ v8f wmma_bf16(v16bf a, v16bf b, v8f c) {
    return __builtin_amdgcn_wmma_f32_16x16x32_bf16(
        false, a, false, b, (short)0, c, false, false);
}

// ---------------------------------------------------------------------------
// Pack kernel: W_hh (1024x256), W2 (256x256) f32 -> bf16 in WMMA-B per-lane
// order. Tile (nt,kt): N = nt*16..+15, K = kt*32..+31. Lane l holds
// n = nt*16+(l&15), K run = kt*32 + (l>>4)*16 .. +15 (contiguous k order chosen here: lanes 0-15
// K=0..15, lanes 16-31 K=16..31). Also bsum = b_ih + b_hh.
// ---------------------------------------------------------------------------
__global__ void pack_weights(const float* __restrict__ Whh,
                             const float* __restrict__ W2,
                             const float* __restrict__ b_ih,
                             const float* __restrict__ b_hh,
                             ushort_t* __restrict__ whh_pk,
                             ushort_t* __restrict__ w2_pk,
                             float* __restrict__ bsum)
{
    const int WHH_ITEMS = 512 * 32;   // (64 n-tiles * 8 k-tiles) * 32 lanes
    const int W2_ITEMS  = 128 * 32;   // (16 n-tiles * 8 k-tiles) * 32 lanes
    int t = blockIdx.x * blockDim.x + threadIdx.x;
    if (t < WHH_ITEMS) {
        int tile = t >> 5, lane = t & 31;
        int nt = tile >> 3, kt = tile & 7;
        int n  = nt * 16 + (lane & 15);
        int k0 = kt * 32 + (lane >> 4) * 16;
        ushort_t* dst = whh_pk + (size_t)tile * 512 + lane * 16;
        const float* src = Whh + (size_t)n * HIDDEN + k0;
        #pragma unroll
        for (int i = 0; i < 16; ++i) dst[i] = f2bf(src[i]);
    } else if (t < WHH_ITEMS + W2_ITEMS) {
        int tt = t - WHH_ITEMS;
        int tile = tt >> 5, lane = tt & 31;
        int nt = tile >> 3, kt = tile & 7;
        int n  = nt * 16 + (lane & 15);
        int k0 = kt * 32 + (lane >> 4) * 16;
        ushort_t* dst = w2_pk + (size_t)tile * 512 + lane * 16;
        const float* src = W2 + (size_t)n * HIDDEN + k0;
        #pragma unroll
        for (int i = 0; i < 16; ++i) dst[i] = f2bf(src[i]);
    } else if (t < WHH_ITEMS + W2_ITEMS + 4 * HIDDEN) {
        int i = t - WHH_ITEMS - W2_ITEMS;
        bsum[i] = b_ih[i] + b_hh[i];
    }
}

// ---------------------------------------------------------------------------
// Main recurrence: 32 WGs x 16 batch rows, 1024 threads (32 waves) each.
// Weights are STREAMED from the shared L2-resident packed arrays every step
// (640KB hot set, identical for all WGs). An opaque zero offset re-defined
// each iteration blocks LICM so the loads stay in-loop as normal global
// loads (clause + partial-wait pipelined) instead of hoist-then-spill.
//   wave w: gate N-tiles { (2*(w>>4)+n2)*16 + (w&15) }
//           logit N-tile (w>>1), K-half (w&1)
// ---------------------------------------------------------------------------
__global__ void __launch_bounds__(NTHREADS, 1)
lstm_decoder_kernel(const float* __restrict__ enc,
                    const float* __restrict__ b2,
                    const ushort_t* __restrict__ whh_pk,
                    const ushort_t* __restrict__ w2_pk,
                    const float* __restrict__ bsum,
                    float* __restrict__ out)
{
    extern __shared__ __align__(16) char smem[];
    float*    lds_gates = (float*)smem;                    // 16*1024 f32 = 64KB
    float*    lds_part  = (float*)(smem + 64 * 1024);      // 2*16*256 f32 = 32KB
    ushort_t* lds_h     = (ushort_t*)(smem + 96 * 1024);   // 16*256 bf16 = 8KB
    float*    lds_redA  = (float*)(smem + 104 * 1024);     // 16*64 = 4KB (max)
    float*    lds_redB  = (float*)(smem + 108 * 1024);     // 16*64 = 4KB (sum)
    // total dynamic LDS: 112KB (WGP budget is 320KB)

    const int tid   = threadIdx.x;
    const int lane  = tid & 31;
    const int w     = tid >> 5;        // wave 0..31
    const int lrow  = lane & 15;
    const int lhalf = lane >> 4;
    const int b0    = blockIdx.x * M_TILE;

    const int ht   = w & 15;           // hidden tile 0..15
    const int gp   = w >> 4;           // gate pair: gates {2gp, 2gp+1}
    const int l_nt = w >> 1;           // logit N-tile
    const int l_kh = w & 1;            // logit K half

    // per-wave tile offsets (u4 units; tile stride = 1KB = 64 u4)
    const int g0_off = (((2 * gp + 0) * 16 + ht) * 8) * 64 + lane * 2;
    const int g1_off = (((2 * gp + 1) * 16 + ht) * 8) * 64 + lane * 2;
    const int l_off  = ((l_nt * 8 + l_kh * 4)) * 64 + lane * 2;

    float gbias[2];
    #pragma unroll
    for (int n2 = 0; n2 < 2; ++n2)
        gbias[n2] = bsum[(2 * gp + n2) * HIDDEN + ht * 16 + lrow];

    // ---- elementwise-thread mapping: thread = (row, 4 columns) ----
    const int erow = tid >> 6;               // 0..15
    const int ecol = (tid & 63) * 4;         // 0..252
    f4 c_reg = *(const f4*)&enc[(size_t)(b0 + erow) * (T_ENC * HIDDEN)
                                + (T_ENC - 1) * HIDDEN + ecol];
    f4 b2v = *(const f4*)&b2[ecol];

    // h0 = 0
    for (int i = tid; i < M_TILE * HIDDEN / 2; i += NTHREADS)
        ((unsigned*)lds_h)[i] = 0u;
    __syncthreads();

    for (int t = 0; t < SEQLEN; ++t) {
        // Opaque zero: blocks LICM of the weight loads (they must re-issue
        // each step, staying as pipelined global loads; no hoist-then-spill).
        unsigned toff = 0;
        asm volatile("" : "+s"(toff));
        const u4* whh_base = (const u4*)whh_pk + toff;
        const u4* w2_base  = (const u4*)w2_pk  + toff;

        // ---------- gates = h_{t-1} @ Whh^T (stream B from L2) ----------
        v8f acc0 = {}, acc1 = {};
        #pragma unroll
        for (int kt = 0; kt < 8; ++kt) {
            BF16Tile bt0, bt1, a;
            bt0.q[0] = whh_base[g0_off + kt * 64];
            bt0.q[1] = whh_base[g0_off + kt * 64 + 1];
            bt1.q[0] = whh_base[g1_off + kt * 64];
            bt1.q[1] = whh_base[g1_off + kt * 64 + 1];
            int k0 = kt * 32 + lhalf * 16;                // contiguous A striping (matches B)
            a.q[0] = *(const u4*)&lds_h[lrow * HIDDEN + k0];
            a.q[1] = *(const u4*)&lds_h[lrow * HIDDEN + k0 + 8];
            acc0 = wmma_bf16(a.v, bt0.v, acc0);
            acc1 = wmma_bf16(a.v, bt1.v, acc1);
        }
        // store biased pre-activations to LDS
        #pragma unroll
        for (int n2 = 0; n2 < 2; ++n2) {
            int col  = (2 * gp + n2) * HIDDEN + ht * 16 + lrow;
            float bb = gbias[n2];
            #pragma unroll
            for (int r = 0; r < 8; ++r) {
                int m = r + 8 * lhalf;
                lds_gates[m * 1024 + col] = (n2 ? acc1[r] : acc0[r]) + bb;
            }
        }
        __syncthreads();

        // ---------- LSTM cell, per-thread natural layout ----------
        {
            f4 gi = *(const f4*)&lds_gates[erow * 1024 + 0 * HIDDEN + ecol];
            f4 gf = *(const f4*)&lds_gates[erow * 1024 + 1 * HIDDEN + ecol];
            f4 gg = *(const f4*)&lds_gates[erow * 1024 + 2 * HIDDEN + ecol];
            f4 go = *(const f4*)&lds_gates[erow * 1024 + 3 * HIDDEN + ecol];
            float hv[4];
            #pragma unroll
            for (int i = 0; i < 4; ++i) {
                float si = fast_sigmoid(gi[i]);
                float sf = fast_sigmoid(gf[i]);
                float so = fast_sigmoid(go[i]);
                float tg = fast_tanh(gg[i]);
                float cn = sf * c_reg[i] + si * tg;
                c_reg[i] = cn;
                hv[i] = so * fast_tanh(cn);
            }
            u2 hp;
            hp.x = (unsigned)f2bf(hv[0]) | ((unsigned)f2bf(hv[1]) << 16);
            hp.y = (unsigned)f2bf(hv[2]) | ((unsigned)f2bf(hv[3]) << 16);
            *(u2*)&lds_h[erow * HIDDEN + ecol] = hp;
        }
        __syncthreads();

        // ---------- logits partials = h_t @ W2^T (half-K per wave) ----------
        v8f lac = {};
        #pragma unroll
        for (int i = 0; i < 4; ++i) {
            BF16Tile bt, a;
            bt.q[0] = w2_base[l_off + i * 64];
            bt.q[1] = w2_base[l_off + i * 64 + 1];
            int kt = l_kh * 4 + i;
            int k0 = kt * 32 + lhalf * 16;
            a.q[0] = *(const u4*)&lds_h[lrow * HIDDEN + k0];
            a.q[1] = *(const u4*)&lds_h[lrow * HIDDEN + k0 + 8];
            lac = wmma_bf16(a.v, bt.v, lac);
        }
        #pragma unroll
        for (int r = 0; r < 8; ++r) {
            int m = r + 8 * lhalf;
            lds_part[l_kh * (M_TILE * OUT_N) + m * OUT_N + l_nt * 16 + lrow] = lac[r];
        }
        __syncthreads();

        // ---------- softmax (row = erow, 4 cols per thread) ----------
        {
            f4 v0 = *(const f4*)&lds_part[erow * OUT_N + ecol];
            f4 v1 = *(const f4*)&lds_part[M_TILE * OUT_N + erow * OUT_N + ecol];
            f4 vals = v0 + v1 + b2v;
            float mx = fmaxf(fmaxf(vals[0], vals[1]), fmaxf(vals[2], vals[3]));
            lds_redA[erow * 64 + (tid & 63)] = mx;
            __syncthreads();
            float rm = -3.0e38f;
            #pragma unroll
            for (int i = 0; i < 16; ++i) {
                f4 q = *(const f4*)&lds_redA[erow * 64 + i * 4];
                rm = fmaxf(rm, fmaxf(fmaxf(q[0], q[1]), fmaxf(q[2], q[3])));
            }
            f4 e;
            float s = 0.f;
            #pragma unroll
            for (int i = 0; i < 4; ++i) { e[i] = __expf(vals[i] - rm); s += e[i]; }
            lds_redB[erow * 64 + (tid & 63)] = s;
            __syncthreads();
            float ss = 0.f;
            #pragma unroll
            for (int i = 0; i < 16; ++i) {
                f4 q = *(const f4*)&lds_redB[erow * 64 + i * 4];
                ss += (q[0] + q[1]) + (q[2] + q[3]);
            }
            float rinv = 1.f / ss;
            *(f4*)&out[((size_t)(b0 + erow) * SEQLEN + t) * OUT_N + ecol] = e * rinv;
            __syncthreads();   // protect lds_redA/B & lds_gates for next step
        }
    }
}

// ---------------------------------------------------------------------------
extern "C" void kernel_launch(void* const* d_in, const int* in_sizes, int n_in,
                              void* d_out, int out_size, void* d_ws, size_t ws_size,
                              hipStream_t stream)
{
    // setup_inputs order: input, encoder_output, W_ih, b_ih, W_hh, b_hh, W2, b2
    const float* enc  = (const float*)d_in[1];
    const float* b_ih = (const float*)d_in[3];
    const float* Whh  = (const float*)d_in[4];
    const float* b_hh = (const float*)d_in[5];
    const float* W2   = (const float*)d_in[6];
    const float* b2   = (const float*)d_in[7];
    float* out = (float*)d_out;

    // Workspace: 512KB packed W_hh bf16 | 128KB packed W2 bf16 | 4KB bsum f32
    ushort_t* whh_pk = (ushort_t*)d_ws;
    ushort_t* w2_pk  = (ushort_t*)((char*)d_ws + 512 * 1024);
    float*    bsum   = (float*)   ((char*)d_ws + 640 * 1024);

    const int pack_items = 512 * 32 + 128 * 32 + 4 * HIDDEN;   // 21504
    pack_weights<<<(pack_items + 255) / 256, 256, 0, stream>>>(
        Whh, W2, b_ih, b_hh, whh_pk, w2_pk, bsum);

    const size_t lds_bytes = 112 * 1024;
    lstm_decoder_kernel<<<NBLOCKS, NTHREADS, lds_bytes, stream>>>(
        enc, b2, whh_pk, w2_pk, bsum, out);
}
